// MoIE_67087389163777
// MI455X (gfx1250) — matrix operand order repaired and checked by reading, not compiled, not run on hardware
//
#include <hip/hip_runtime.h>
#include <hip/hip_bf16.h>

typedef __attribute__((ext_vector_type(16))) _Float16 v16h;
typedef __attribute__((ext_vector_type(8)))  float    v8f;

#define BQ 32768
#define DQ 256
#define KQ 8

#define CHUNK_HALFS 16384      // one (expert, 32-K-row) chunk: hi[8192] + lo[8192] f16
#define CHUNK_BYTES (CHUNK_HALFS * 2)

// ---------------------------------------------------------------------------
// Gate: alpha = softmax(x @ gate_W + gate_b), one thread per row.
// ---------------------------------------------------------------------------
__global__ __launch_bounds__(256) void moie_gate(const float* __restrict__ x,
                                                 const float* __restrict__ gW,
                                                 const float* __restrict__ gb,
                                                 float* __restrict__ alpha) {
    const int b = blockIdx.x * blockDim.x + threadIdx.x;
    float z[KQ];
#pragma unroll
    for (int k = 0; k < KQ; ++k) z[k] = gb[k];
    const float* xr = x + (size_t)b * DQ;
    for (int i = 0; i < DQ; ++i) {
        const float xv = xr[i];
#pragma unroll
        for (int k = 0; k < KQ; ++k) z[k] = fmaf(xv, gW[i * KQ + k], z[k]);
    }
    float m = z[0];
#pragma unroll
    for (int k = 1; k < KQ; ++k) m = fmaxf(m, z[k]);
    float s = 0.f;
#pragma unroll
    for (int k = 0; k < KQ; ++k) { z[k] = __expf(z[k] - m); s += z[k]; }
    const float inv = 1.f / s;
#pragma unroll
    for (int k = 0; k < KQ; ++k) alpha[(size_t)b * KQ + k] = z[k] * inv;
}

// ---------------------------------------------------------------------------
// Pre-pack one layer's W (f32 [8][256][256]) into hi/lo f16 B-fragment chunks:
// dst[chunk(k*8+ks)][ hi: (nt*32+r)*16+nn | lo: +8192 ], chunk = 32 KB contiguous.
// Grid: 64 blocks x 512 threads per layer.
// ---------------------------------------------------------------------------
__global__ __launch_bounds__(512) void moie_pack(const float* __restrict__ W,
                                                 _Float16* __restrict__ dst) {
    const int blk = blockIdx.x;            // 0..63 : k = blk>>3, ks = blk&7
    const int k  = blk >> 3;
    const int ks = blk & 7;
    const int t  = threadIdx.x;
    const int nt = t >> 5;                 // 0..15
    const int r  = t & 31;                 // 0..31 (B-fragment lane: col r&15, K half r>>4)
    const float* src = W + (((size_t)k * DQ + ks * 32) * DQ + nt * 16 + (r & 15));
    _Float16* dh = dst + (size_t)blk * CHUNK_HALFS + (nt * 32 + r) * 16;
    _Float16* dl = dh + 8192;
    v16h hi, lo;
#pragma unroll
    for (int nn = 0; nn < 16; ++nn) {
        const float w  = src[(size_t)((nn & 7) + 8 * (r >> 4) + 16 * (nn >> 3)) * DQ];
        const _Float16 wh = (_Float16)w;
        hi[nn] = wh;
        lo[nn] = (_Float16)(w - (float)wh);
    }
    *(v16h*)dh = hi;
    *(v16h*)dl = lo;
}

// ---------------------------------------------------------------------------
// Async staging: copy this thread's 64 B slice of a packed chunk into LDS.
// Same IOFFSET applies to both LDS dest and global source (ISA 10.7/15.18).
// ---------------------------------------------------------------------------
__device__ __forceinline__ void stage_async(const _Float16* __restrict__ gsrc,
                                            _Float16* lds_dst) {
    const unsigned long long ga = (unsigned long long)(uintptr_t)gsrc;
    const unsigned la = (unsigned)(uintptr_t)lds_dst;   // flat LDS addr[31:0] = LDS offset
    asm volatile(
        "global_load_async_to_lds_b128 %0, %1, off\n\t"
        "global_load_async_to_lds_b128 %0, %1, off offset:16\n\t"
        "global_load_async_to_lds_b128 %0, %1, off offset:32\n\t"
        "global_load_async_to_lds_b128 %0, %1, off offset:48"
        :: "v"(la), "v"(ga) : "memory");
}

// ---------------------------------------------------------------------------
// One MoIE layer: Hout[b,:] = act( sum_k alpha[b,k] * (Hin[b,:] @ W[k] + bias[k]) )
// f32 via f16 hi/lo split: A*B = Ah*Bh + Ah*Bl + Al*Bh (f32 accum).
// WG: 512 thr = 16 waves; 8 M-tiles (16 rows) x 2 N-halves (128 cols).
// Double-buffered async LDS staging, one barrier per chunk.
// ---------------------------------------------------------------------------
template <bool RELU>
__global__ __launch_bounds__(512) void moie_layer(const float* __restrict__ Hin,
                                                  float* __restrict__ Hout,
                                                  const _Float16* __restrict__ Wp,  // [64][16384]
                                                  const float* __restrict__ bias,   // [8][256]
                                                  const float* __restrict__ alpha   // [B][8]
) {
    __shared__ __align__(32) _Float16 sB[2][CHUNK_HALFS];  // 2 x 32 KB

    const int tid     = threadIdx.x;
    const int lane    = tid & 31;
    const int wave    = tid >> 5;
    const int mTile   = wave >> 1;       // 0..7
    const int nHalf   = wave & 1;        // 0..1
    const int laneRow = lane & 15;
    const int hiHalf  = (lane >> 4) & 1;
    const int rowBase = blockIdx.x * 128 + mTile * 16;

    // kick off chunk 0 into buffer 0 before the (long) A-fragment build
    stage_async(Wp + (size_t)tid * 32, &sB[0][tid * 32]);

    // ---- Build A fragments (16 rows x 256 K) once: hi + lo f16 splits ----
    v16h Ahi[8], Alo[8];
    {
        const float* arow = Hin + (size_t)(rowBase + laneRow) * DQ;
#pragma unroll
        for (int ks = 0; ks < 8; ++ks) {
#pragma unroll
            for (int v = 0; v < 8; ++v) {
                const int kb = ks * 32 + (v >> 2) * 16 + hiHalf * 8 + (v & 3) * 2;
                const float2 p = *(const float2*)(arow + kb);
                const _Float16 h0 = (_Float16)p.x;
                const _Float16 h1 = (_Float16)p.y;
                Ahi[ks][2 * v]     = h0;
                Ahi[ks][2 * v + 1] = h1;
                Alo[ks][2 * v]     = (_Float16)(p.x - (float)h0);
                Alo[ks][2 * v + 1] = (_Float16)(p.y - (float)h1);
            }
        }
    }

    v8f acc[8];
#pragma unroll
    for (int j = 0; j < 8; ++j) acc[j] = (v8f){0.f, 0.f, 0.f, 0.f, 0.f, 0.f, 0.f, 0.f};

    for (int k = 0; k < KQ; ++k) {
        v8f tmp[8];
#pragma unroll
        for (int j = 0; j < 8; ++j) tmp[j] = (v8f){0.f, 0.f, 0.f, 0.f, 0.f, 0.f, 0.f, 0.f};

#pragma unroll
        for (int ks = 0; ks < 8; ++ks) {
            const int c   = k * 8 + ks;
            const int cur = ks & 1;            // == c & 1 (k*8 is even)

            // my async fills of buf[cur] (issued last iteration) are done:
            asm volatile("s_wait_asynccnt 0x0" ::: "memory");
            __syncthreads();                   // everyone's fills visible; buf[cur^1] free

            if (c < 63) {                      // prefetch next chunk into the other buffer
                stage_async(Wp + (size_t)(c + 1) * CHUNK_HALFS + tid * 32,
                            &sB[cur ^ 1][tid * 32]);
            }

#pragma unroll
            for (int j = 0; j < 8; ++j) {
                const int nt = nHalf * 8 + j;
                const v16h bh = *(const v16h*)&sB[cur][(nt * 32 + lane) * 16];
                const v16h bl = *(const v16h*)&sB[cur][8192 + (nt * 32 + lane) * 16];
                tmp[j] = __builtin_amdgcn_wmma_f32_16x16x32_f16(
                    false, Ahi[ks], false, bh, (short)0, tmp[j], false, false);
                tmp[j] = __builtin_amdgcn_wmma_f32_16x16x32_f16(
                    false, Ahi[ks], false, bl, (short)0, tmp[j], false, false);
                tmp[j] = __builtin_amdgcn_wmma_f32_16x16x32_f16(
                    false, Alo[ks], false, bh, (short)0, tmp[j], false, false);
            }
        }

        // ---- epilogue: acc += alpha[b,k] * (tmp + bias[k]) in full f32 ----
        float areg[8];
#pragma unroll
        for (int r = 0; r < 8; ++r)
            areg[r] = alpha[(size_t)(rowBase + r + 8 * hiHalf) * KQ + k];
#pragma unroll
        for (int j = 0; j < 8; ++j) {
            const int nt = nHalf * 8 + j;
            const float bv = bias[k * DQ + nt * 16 + laneRow];
#pragma unroll
            for (int r = 0; r < 8; ++r)
                acc[j][r] = fmaf(areg[r], tmp[j][r] + bv, acc[j][r]);
        }
    }

    // ---- activation + store (C layout: vgpr r -> row, lane%16 -> col) ----
#pragma unroll
    for (int j = 0; j < 8; ++j) {
        const int col = (nHalf * 8 + j) * 16 + laneRow;
#pragma unroll
        for (int r = 0; r < 8; ++r) {
            const int row = rowBase + r + 8 * hiHalf;
            float v = acc[j][r];
            if (RELU) v = fmaxf(v, 0.f);
            Hout[(size_t)row * DQ + col] = v;
        }
    }
}

// ---------------------------------------------------------------------------
extern "C" void kernel_launch(void* const* d_in, const int* in_sizes, int n_in,
                              void* d_out, int out_size, void* d_ws, size_t ws_size,
                              hipStream_t stream) {
    const float* x       = (const float*)d_in[0];
    const float* gate_W  = (const float*)d_in[1];
    const float* gate_b  = (const float*)d_in[2];
    const float* block_W = (const float*)d_in[3]; // [4][8][256][256]
    const float* block_b = (const float*)d_in[4]; // [4][8][256]
    const float* out_W   = (const float*)d_in[5]; // [8][256][256]
    const float* out_b   = (const float*)d_in[6]; // [8][256]
    float* out = (float*)d_out;

    // ws layout: hB (32MB) | alpha (1MB) | packed W, 5 layers x 2MB (10MB)
    float*    hB    = (float*)d_ws;
    float*    alpha = (float*)((char*)d_ws + (size_t)BQ * DQ * sizeof(float));
    _Float16* Wpk   = (_Float16*)((char*)alpha + (size_t)BQ * KQ * sizeof(float));
    const size_t WPK_L = (size_t)64 * CHUNK_HALFS;  // halfs per packed layer

    const size_t WL = (size_t)KQ * DQ * DQ;
    const size_t BL = (size_t)KQ * DQ;

    // pre-pack weights (tiny: 10MB once) + gate in parallel-ish on the stream
    for (int l = 0; l < 4; ++l)
        moie_pack<<<64, 512, 0, stream>>>(block_W + l * WL, Wpk + l * WPK_L);
    moie_pack<<<64, 512, 0, stream>>>(out_W, Wpk + 4 * WPK_L);
    moie_gate<<<BQ / 256, 256, 0, stream>>>(x, gate_W, gate_b, alpha);

    dim3 g(BQ / 128), b(512);
    // ping-pong: d_out <-> d_ws (no intra-kernel aliasing)
    moie_layer<true ><<<g, b, 0, stream>>>(x,   out, Wpk + 0 * WPK_L, block_b + 0 * BL, alpha);
    moie_layer<true ><<<g, b, 0, stream>>>(out, hB,  Wpk + 1 * WPK_L, block_b + 1 * BL, alpha);
    moie_layer<true ><<<g, b, 0, stream>>>(hB,  out, Wpk + 2 * WPK_L, block_b + 2 * BL, alpha);
    moie_layer<true ><<<g, b, 0, stream>>>(out, hB,  Wpk + 3 * WPK_L, block_b + 3 * BL, alpha);
    moie_layer<false><<<g, b, 0, stream>>>(hB,  out, Wpk + 4 * WPK_L, out_b,            alpha);
}
